// GroupedQueryAttention_25194278158404
// MI455X (gfx1250) — matrix operand order repaired and checked by reading, not compiled, not run on hardware
//
#include <hip/hip_runtime.h>
#include <hip/hip_bf16.h>

// ---------------------------------------------------------------------------
// GQA + ALiBi attention for MI455X (gfx1250).
// All matmuls via v_wmma_f32_16x16x32_bf16 (wave32 WMMA), fp32 softmax stats.
// K/V tiles staged in LDS via gfx1250 async-to-LDS DMA when available.
// B=4, S=2048, D=1024, H=16, KV=4, HD=64, GROUPS=4
// ---------------------------------------------------------------------------

typedef __attribute__((ext_vector_type(16))) __bf16 bf16x16;
typedef __attribute__((ext_vector_type(8)))  __bf16 bf16x8;
typedef __attribute__((ext_vector_type(8)))  float  f32x8;
typedef __attribute__((ext_vector_type(4)))  int    i32x4;

#if __has_builtin(__builtin_amdgcn_global_load_async_to_lds_b128) && \
    __has_builtin(__builtin_amdgcn_s_wait_asynccnt)
#define USE_ASYNC_LDS 1
#else
#define USE_ASYNC_LDS 0
#endif

static __device__ __forceinline__ f32x8 wmma_bf16(bf16x16 a, bf16x16 b, f32x8 c) {
    return __builtin_amdgcn_wmma_f32_16x16x32_bf16(false, a, false, b,
                                                   (short)0, c, false, false);
}

#if USE_ASYNC_LDS
// Signature (upstream clang): (int4 addrspace(1)* gsrc, int4 addrspace(3)* ldst,
//                              imm int offset, imm int cpol)
static __device__ __forceinline__ void async_ld16(__bf16* lds, const __bf16* g) {
    __attribute__((address_space(1))) i32x4* gp =
        (__attribute__((address_space(1))) i32x4*)(__bf16*)g;
    __attribute__((address_space(3))) i32x4* lp =
        (__attribute__((address_space(3))) i32x4*)lds;
    __builtin_amdgcn_global_load_async_to_lds_b128(gp, lp, 0, 0);
}
#endif

// A-fragment (16xK tile, row-major, ld elements between rows), K-offset k0.
// Lane L: row = L&15; element e: K = k0 + 16*(e>=8) + 8*(L>=16) + (e&7)
static __device__ __forceinline__ bf16x16
load_fragA(const __bf16* base, int ld, int k0) {
    int lane = threadIdx.x & 31;
    const __bf16* p = base + (size_t)(lane & 15) * ld + k0 + ((lane >> 4) & 1) * 8;
    bf16x8 lo = *(const bf16x8*)(p);
    bf16x8 hi = *(const bf16x8*)(p + 16);
    return __builtin_shufflevector(lo, hi, 0,1,2,3,4,5,6,7,8,9,10,11,12,13,14,15);
}

// B-fragment from "Bt" storage: row n holds the K-contiguous column n of B.
// Lane L: n = L&15; element e: K = k0 + 16*(e>=8) + 8*(L>=16) + (e&7)
static __device__ __forceinline__ bf16x16
load_fragB(const __bf16* base, int ld, int k0) {
    int lane = threadIdx.x & 31;
    const __bf16* p = base + (size_t)(lane & 15) * ld + k0 + ((lane >> 4) & 1) * 8;
    bf16x8 lo = *(const bf16x8*)(p);
    bf16x8 hi = *(const bf16x8*)(p + 16);
    return __builtin_shufflevector(lo, hi, 0,1,2,3,4,5,6,7,8,9,10,11,12,13,14,15);
}

// ---------------------------------------------------------------------------
// Prep kernels
// ---------------------------------------------------------------------------
__global__ void cvt_f32_bf16(const float* __restrict__ src,
                             __bf16* __restrict__ dst, int n) {
    int i = blockIdx.x * 256 + threadIdx.x;
    if (i < n) dst[i] = (__bf16)src[i];
}

// Pack [Wq|Wk|Wv] transposed: Wt[n][k], n in [0,1536), k in [0,1024)
__global__ void pack_wqkv(const float* __restrict__ Wq,
                          const float* __restrict__ Wk,
                          const float* __restrict__ Wv,
                          __bf16* __restrict__ Wt) {
    int idx = blockIdx.x * 256 + threadIdx.x;   // over 1536*1024
    int k = idx & 1023;
    int n = idx >> 10;
    float v;
    if (n < 1024)       v = Wq[(size_t)k * 1024 + n];
    else if (n < 1280)  v = Wk[(size_t)k * 256 + (n - 1024)];
    else                v = Wv[(size_t)k * 256 + (n - 1280)];
    Wt[idx] = (__bf16)v;
}

__global__ void pack_wo(const float* __restrict__ Wo, __bf16* __restrict__ Wt) {
    int idx = blockIdx.x * 256 + threadIdx.x;   // over 1024*1024
    int k = idx & 1023;
    int n = idx >> 10;
    Wt[idx] = (__bf16)Wo[(size_t)k * 1024 + n];
}

// Vt[b][kv][hd][s] <- qkv[b*2048+s][1280 + kv*64 + hd]
__global__ void v_transpose(const __bf16* __restrict__ qkv,
                            __bf16* __restrict__ Vt) {
    int idx = blockIdx.x * 256 + threadIdx.x;   // over 4*4*64*2048
    int s    = idx & 2047;
    int rest = idx >> 11;
    int hd   = rest & 63;
    int kvb  = rest >> 6;                        // b*4+kv
    int kv   = kvb & 3;
    int b    = kvb >> 2;
    Vt[idx] = qkv[(size_t)(b * 2048 + s) * 1536 + 1280 + kv * 64 + hd];
}

// ---------------------------------------------------------------------------
// bf16 WMMA GEMM: C[M,N] = A[M,K] @ B[K,N] ; B given transposed (Bt[n][k]).
// 8 waves/block: 2 (M) x 4 (N); wave tile 32x64 (2 A-frags x 4 B-frags =
// 8 wmma per 12 x 16B loads); block tile 64x256.
// Output to bf16 (scale applied to cols < scaleNcut) or fp32.
// ---------------------------------------------------------------------------
__global__ __launch_bounds__(256) void gemm_bf16_wmma(
    const __bf16* __restrict__ A, int lda,
    const __bf16* __restrict__ Bt, int ldb,
    int Kdim,
    __bf16* __restrict__ outBf, float* __restrict__ outF, int ldo,
    float scale, int scaleNcut) {
    int wave = threadIdx.x >> 5;
    int lane = threadIdx.x & 31;
    int wm = wave & 1, wn = wave >> 1;
    int mBase = blockIdx.y * 64 + wm * 32;
    int nBase = blockIdx.x * 256 + wn * 64;

    const __bf16* At0 = A + (size_t)mBase * lda;
    const __bf16* At1 = A + (size_t)(mBase + 16) * lda;
    f32x8 acc[2][4] = {};

    for (int k0 = 0; k0 < Kdim; k0 += 32) {
        bf16x16 a0 = load_fragA(At0, lda, k0);
        bf16x16 a1 = load_fragA(At1, lda, k0);
#pragma unroll
        for (int t = 0; t < 4; ++t) {
            bf16x16 b = load_fragB(Bt + (size_t)(nBase + t * 16) * ldb, ldb, k0);
            acc[0][t] = wmma_bf16(a0, b, acc[0][t]);
            acc[1][t] = wmma_bf16(a1, b, acc[1][t]);
        }
    }

    int rowOff = ((lane >> 4) & 1) * 8;
#pragma unroll
    for (int i = 0; i < 2; ++i) {
#pragma unroll
        for (int t = 0; t < 4; ++t) {
            int col = nBase + t * 16 + (lane & 15);
            float s = (col < scaleNcut) ? scale : 1.0f;
#pragma unroll
            for (int r = 0; r < 8; ++r) {
                size_t row = (size_t)(mBase + i * 16 + rowOff + r);
                float v = acc[i][t][r] * s;
                if (outF) outF[row * ldo + col] = v;
                else      outBf[row * ldo + col] = (__bf16)v;
            }
        }
    }
}

// ---------------------------------------------------------------------------
// Flash-style GQA attention with ALiBi.
// grid.x = b*16+h (64), grid.y = q-block (16).  8 waves/block, one 16-query
// tile per wave.  Keys streamed in 32-wide chunks; K/V tiles double-buffered
// in LDS (async-to-LDS DMA when available, else register-staged).
// P is per-wave LDS: same-wave LDS ops are in-order, so the P write->read
// turnaround needs no barrier.  Exactly one barrier per chunk:
//   iter c:  [issue loads c+1 -> buf nb] ; QK(buf) ; softmax ; P ; PV(buf) ;
//            [land c+1 in buf nb] ; wait-async ; barrier
// All reads of a buffer complete before the barrier of their own iteration;
// the next write of that buffer is one barrier later.
// ---------------------------------------------------------------------------
__global__ __launch_bounds__(256) void attn_wmma(
    const __bf16* __restrict__ qkv,  // [B*S, 1536]  (Q pre-scaled by 0.125)
    const __bf16* __restrict__ Vt,   // [B*KV*64, 2048]
    __bf16* __restrict__ ctx) {      // [B*S, 1024]
    const int S = 2048, LDQ = 1536;
    int bh = blockIdx.x;
    int b  = bh >> 4, h = bh & 15;
    int kv = h >> 2;                        // h / GROUPS
    int wave = threadIdx.x >> 5, lane = threadIdx.x & 31;
    int tid  = threadIdx.x;
    int qtile = blockIdx.y * 8 + wave;      // 0..127
    int qRow  = qtile * 16;

    const __bf16* Qbase = qkv + (size_t)(b * S + qRow) * LDQ + h * 64;
    const __bf16* Kbase = qkv + (size_t)(b * S) * LDQ + 1024 + kv * 64;
    const __bf16* Vbase = Vt + (size_t)((b * 4 + kv) * 64) * S;  // row=hd, ld=S

    float slope = exp2f(-0.5f * (float)(h + 1));

    bf16x16 qa0 = load_fragA(Qbase, LDQ, 0);
    bf16x16 qa1 = load_fragA(Qbase, LDQ, 32);

    // LDS: K tile [32 keys][64 hd] x2, V tile [64 hd][32 keys] x2, P per-wave.
    __shared__ alignas(16) __bf16 Kbuf[2][32][64];   // 8 KB
    __shared__ alignas(16) __bf16 Vbuf[2][64][32];   // 8 KB
    __shared__ alignas(16) __bf16 Plds[8][16][32];   // 8 KB

    // Per-thread cooperative-copy assignment (256 threads, 16B units).
    int krow = tid >> 3, kcc = tid & 7;     // 32 rows x 8 chunks
    int vrow = tid >> 2, vcc = tid & 3;     // 64 rows x 4 chunks
    const __bf16* kTB = Kbase + (size_t)krow * LDQ + kcc * 8;
    const __bf16* vTB = Vbase + (size_t)vrow * S + vcc * 8;

    float m[8], l[8];
#pragma unroll
    for (int r = 0; r < 8; ++r) { m[r] = -1e30f; l[r] = 0.0f; }
    f32x8 o0 = {}, o1 = {}, o2 = {}, o3 = {};

    int kCol   = lane & 15;
    int rowOff = ((lane >> 4) & 1) * 8;
    float qAbsBase = (float)(qRow + rowOff);

    // Prologue: chunk 0 into buffer 0.
#if USE_ASYNC_LDS
    async_ld16(&Kbuf[0][krow][kcc * 8], kTB);
    async_ld16(&Vbuf[0][vrow][vcc * 8], vTB);
    __builtin_amdgcn_s_wait_asynccnt(0);
#else
    *(uint4*)&Kbuf[0][krow][kcc * 8] = *(const uint4*)(kTB);
    *(uint4*)&Vbuf[0][vrow][vcc * 8] = *(const uint4*)(vTB);
#endif
    __syncthreads();

    for (int c = 0; c < 64; ++c) {
        int buf = c & 1, nb = buf ^ 1;
        int key0 = c * 32;

        // Kick off next chunk's K/V transfer into the other buffer.
#if USE_ASYNC_LDS
        if (c < 63) {
            async_ld16(&Kbuf[nb][krow][kcc * 8], kTB + (size_t)(key0 + 32) * LDQ);
            async_ld16(&Vbuf[nb][vrow][vcc * 8], vTB + (key0 + 32));
        }
#else
        uint4 kreg, vreg;
        if (c < 63) {
            kreg = *(const uint4*)(kTB + (size_t)(key0 + 32) * LDQ);
            vreg = *(const uint4*)(vTB + (key0 + 32));
        }
#endif
        if (c + 2 < 64) {
            __builtin_prefetch(kTB + (size_t)(key0 + 64) * LDQ, 0, 1);
            __builtin_prefetch(vTB + (key0 + 64), 0, 1);
        }

        // ---- scores: two 16-key tiles, K-dim = 64 (2 wmma each), from LDS --
        f32x8 s0 = {}, s1 = {};
        s0 = wmma_bf16(qa0, load_fragB(&Kbuf[buf][0][0],  64, 0),  s0);
        s0 = wmma_bf16(qa1, load_fragB(&Kbuf[buf][0][0],  64, 32), s0);
        s1 = wmma_bf16(qa0, load_fragB(&Kbuf[buf][16][0], 64, 0),  s1);
        s1 = wmma_bf16(qa1, load_fragB(&Kbuf[buf][16][0], 64, 32), s1);

        // ---- ALiBi + per-row chunk max ----
        float mx[8];
#pragma unroll
        for (int r = 0; r < 8; ++r) {
            float qa = qAbsBase + (float)r;
            float v0 = s0[r] + slope * ((float)(key0 + kCol) - qa);
            float v1 = s1[r] + slope * ((float)(key0 + 16 + kCol) - qa);
            s0[r] = v0; s1[r] = v1;
            mx[r] = fmaxf(v0, v1);
        }
#pragma unroll
        for (int mask = 1; mask < 16; mask <<= 1) {
#pragma unroll
            for (int r = 0; r < 8; ++r)
                mx[r] = fmaxf(mx[r], __shfl_xor(mx[r], mask, 32));
        }

        // ---- online softmax update; write P (bf16) into per-wave LDS ----
#pragma unroll
        for (int r = 0; r < 8; ++r) {
            float mnew = fmaxf(m[r], mx[r]);
            float corr = __expf(m[r] - mnew);
            m[r] = mnew;
            l[r] *= corr;
            o0[r] *= corr; o1[r] *= corr; o2[r] *= corr; o3[r] *= corr;
            float p0 = __expf(s0[r] - mnew);
            float p1 = __expf(s1[r] - mnew);
            l[r] += p0 + p1;
            int row = rowOff + r;
            Plds[wave][row][kCol]      = (__bf16)p0;
            Plds[wave][row][16 + kCol] = (__bf16)p1;
        }

        // ---- PV: O[16x64] += P[16x32] @ V[32x64]; same-wave LDS in-order ---
        bf16x16 pa = load_fragA(&Plds[wave][0][0], 32, 0);
        o0 = wmma_bf16(pa, load_fragB(&Vbuf[buf][0][0],  32, 0), o0);
        o1 = wmma_bf16(pa, load_fragB(&Vbuf[buf][16][0], 32, 0), o1);
        o2 = wmma_bf16(pa, load_fragB(&Vbuf[buf][32][0], 32, 0), o2);
        o3 = wmma_bf16(pa, load_fragB(&Vbuf[buf][48][0], 32, 0), o3);

        // ---- land next chunk's K/V, then publish with one barrier ----
#if USE_ASYNC_LDS
        if (c < 63) __builtin_amdgcn_s_wait_asynccnt(0);
#else
        if (c < 63) {
            *(uint4*)&Kbuf[nb][krow][kcc * 8] = kreg;
            *(uint4*)&Vbuf[nb][vrow][vcc * 8] = vreg;
        }
#endif
        __syncthreads();
    }

    // ---- finalize: cross-lane row sums, divide, store bf16 context ----
#pragma unroll
    for (int mask = 1; mask < 16; mask <<= 1) {
#pragma unroll
        for (int r = 0; r < 8; ++r)
            l[r] += __shfl_xor(l[r], mask, 32);
    }
    f32x8 os[4] = {o0, o1, o2, o3};
#pragma unroll
    for (int t = 0; t < 4; ++t) {
        int col = h * 64 + t * 16 + (lane & 15);
#pragma unroll
        for (int r = 0; r < 8; ++r) {
            size_t row = (size_t)(b * S + qRow + rowOff + r);
            ctx[row * 1024 + col] = (__bf16)(os[t][r] / l[r]);
        }
    }
}

// ---------------------------------------------------------------------------
// Host launcher
// ---------------------------------------------------------------------------
extern "C" void kernel_launch(void* const* d_in, const int* in_sizes, int n_in,
                              void* d_out, int out_size, void* d_ws, size_t ws_size,
                              hipStream_t stream) {
    (void)in_sizes; (void)n_in; (void)out_size; (void)ws_size;
    const float* x  = (const float*)d_in[0];
    const float* Wq = (const float*)d_in[1];
    const float* Wk = (const float*)d_in[2];
    const float* Wv = (const float*)d_in[3];
    const float* Wo = (const float*)d_in[4];
    float* out = (float*)d_out;

    const int B = 4, S = 2048, D = 1024;        // tokens = 8192
    const int M = B * S;

    char* ws = (char*)d_ws;
    __bf16* xb    = (__bf16*)(ws);                                   // 16 MB
    __bf16* wqkvt = (__bf16*)(ws + (size_t)16777216);                //  3 MB
    __bf16* wot   = (__bf16*)(ws + (size_t)19922944);                //  2 MB
    __bf16* qkv   = (__bf16*)(ws + (size_t)22020096);                // 24 MB
    __bf16* vt    = (__bf16*)(ws + (size_t)47185920);                //  4 MB
    __bf16* ctx   = (__bf16*)(ws + (size_t)51380224);                // 16 MB

    // 1) convert x to bf16
    cvt_f32_bf16<<<(M * D) / 256, 256, 0, stream>>>(x, xb, M * D);
    // 2) pack transposed weights (bf16)
    pack_wqkv<<<(1536 * 1024) / 256, 256, 0, stream>>>(Wq, Wk, Wv, wqkvt);
    pack_wo<<<(1024 * 1024) / 256, 256, 0, stream>>>(Wo, wot);
    // 3) fused QKV projection: qkv[8192,1536]; Q cols (<1024) scaled by HD^-0.5
    {
        dim3 grid(1536 / 256, M / 64);
        gemm_bf16_wmma<<<grid, 256, 0, stream>>>(xb, D, wqkvt, D, D,
                                                 qkv, nullptr, 1536,
                                                 0.125f, 1024);
    }
    // 4) transpose V into Vt[b][kv][hd][s]
    v_transpose<<<(4 * 4 * 64 * 2048) / 256, 256, 0, stream>>>(qkv, vt);
    // 5) attention -> ctx[8192,1024] bf16
    {
        dim3 grid(64, S / 128);   // (b*16+h, q-block of 8x16 queries)
        attn_wmma<<<grid, 256, 0, stream>>>(qkv, vt, ctx);
    }
    // 6) output projection -> fp32 d_out
    {
        dim3 grid(1024 / 256, M / 64);
        gemm_bf16_wmma<<<grid, 256, 0, stream>>>(ctx, 1024, wot, 1024, 1024,
                                                 nullptr, out, 1024,
                                                 1.0f, 0);
    }
}
